// GlobalMPNNLayer2_14620068675878
// MI455X (gfx1250) — compile-verified, matrix operand order checked by reading, not run on hardware
//
#include <hip/hip_runtime.h>

// ---------------------------------------------------------------------------
// GlobalMPNNLayer2 for MI455X (gfx1250): bf16 WMMA (16x16x32, f32 accum),
// software-pipelined fragment loads, pre-converted bf16 node/global features,
// f32 atomics for segment sums.
// ---------------------------------------------------------------------------

#define NN  50000
#define EE  800000
#define GGG 512
#define HD  128
#define ED  64
#define GD  64
#define HID 128

typedef __attribute__((ext_vector_type(16))) __bf16        v16bf;
typedef __attribute__((ext_vector_type(8)))  float         v8f;
typedef __attribute__((ext_vector_type(8)))  unsigned int  v8u;
typedef __attribute__((ext_vector_type(4)))  unsigned int  v4u;

union Frag { v8u u; v16bf b; };

static __device__ __forceinline__ unsigned short f2bf_bits(float f) {
  unsigned a = __float_as_uint(f);
  a += 0x7fffu + ((a >> 16) & 1u);          // round-to-nearest-even
  return (unsigned short)(a >> 16);
}
static __device__ __forceinline__ unsigned pk2(float lo, float hi) {
  unsigned a = __float_as_uint(lo); a += 0x7fffu + ((a >> 16) & 1u);
  unsigned b = __float_as_uint(hi); b += 0x7fffu + ((b >> 16) & 1u);
  return (a >> 16) | (b & 0xffff0000u);
}
static __device__ __forceinline__ float bf2f(unsigned short s) {
  return __uint_as_float(((unsigned)s) << 16);
}

static __device__ __forceinline__ v8f wmma_bf16(const Frag& A, const Frag& B, v8f C) {
  return __builtin_amdgcn_wmma_f32_16x16x32_bf16(
      false, A.b, false, B.b, (short)0, C, false, false);
}

// A fragment from a bf16 row (global or LDS): two 16B loads, already in
// packed-pair order (word w<4 = elements k0+2w, k0+2w+1; w>=4 = k0+2w+8, k0+2w+9).
static __device__ __forceinline__ Frag afrag_bf(const unsigned short* row, int c0) {
  v4u lo = *(const v4u*)(row + c0);
  v4u hh = *(const v4u*)(row + c0 + 16);
  Frag f;
  f.u = (v8u){lo[0], lo[1], lo[2], lo[3], hh[0], hh[1], hh[2], hh[3]};
  return f;
}

// A fragment from an fp32 row: load 2x8 floats, convert.
static __device__ __forceinline__ void ld_chunk8(const float* __restrict__ p,
                                                 unsigned* __restrict__ o) {
  float4 x = *(const float4*)p;
  float4 y = *(const float4*)(p + 4);
  o[0] = pk2(x.x, x.y); o[1] = pk2(x.z, x.w);
  o[2] = pk2(y.x, y.y); o[3] = pk2(y.z, y.w);
}
static __device__ __forceinline__ Frag afrag_f32(const float* row, int c0) {
  unsigned w[8];
  ld_chunk8(row + c0, w);
  ld_chunk8(row + c0 + 16, w + 4);
  Frag f;
  f.u = (v8u){w[0], w[1], w[2], w[3], w[4], w[5], w[6], w[7]};
  return f;
}

// B fragment: pre-packed weights, 32B per lane per fragment (coalesced).
static __device__ __forceinline__ Frag bfrag(const unsigned short* __restrict__ wp,
                                             int frag, int lane) {
  Frag f;
  f.u = *(const v8u*)(wp + ((size_t)frag * 32 + lane) * 16);
  return f;
}

// ---------------------------------------------------------------------------
// Weight pre-pack: fp32 row-major W[K][N] -> bf16 B-fragment order.
//   lane l: n = nt*16 + (l&15), hi = l>>4
//   vgpr v<4 : k = kt*32 +      hi*8 + 2v     + p
//   vgpr v>=4: k = kt*32 + 16 + hi*8 + 2(v-4) + p
// ---------------------------------------------------------------------------
__global__ __launch_bounds__(256) void pack_kernel(const float* __restrict__ W,
                                                   unsigned short* __restrict__ P,
                                                   int K, int N) {
  int idx = blockIdx.x * 256 + threadIdx.x;
  if (idx >= K * N) return;
  int j  = idx & 15;
  int l  = (idx >> 4) & 31;
  int f  = idx >> 9;
  int NT = N >> 4;
  int kt = f / NT, nt = f % NT;
  int n  = nt * 16 + (l & 15);
  int hi = l >> 4;
  int v  = j >> 1, p = j & 1;
  int k  = kt * 32 + hi * 8 + ((v < 4) ? (2 * v + p) : (16 + 2 * (v - 4) + p));
  P[idx] = f2bf_bits(W[(size_t)k * N + n]);
}

__global__ __launch_bounds__(256) void zero_kernel(float* __restrict__ p, long n) {
  long i = (long)blockIdx.x * 256 + threadIdx.x;
  long s = (long)gridDim.x * 256;
  for (; i < n; i += s) p[i] = 0.0f;
}

__global__ __launch_bounds__(256) void cvt_bf16_kernel(const float* __restrict__ src,
                                                       unsigned short* __restrict__ dst,
                                                       long n) {
  long i = (long)blockIdx.x * 256 + threadIdx.x;
  long s = (long)gridDim.x * 256;
  for (; i < n; i += s) dst[i] = f2bf_bits(src[i]);
}

// ---------------------------------------------------------------------------
// Edge kernel: one wave per 16-edge tile.
// ---------------------------------------------------------------------------
__global__ __launch_bounds__(256) void edge_kernel(
    const unsigned short* __restrict__ h_bf, const int* __restrict__ ei,
    const float* __restrict__ e, const unsigned short* __restrict__ g_bf,
    const int* __restrict__ batch,
    const unsigned short* __restrict__ Wm1, const float* __restrict__ bm1,
    const unsigned short* __restrict__ Wm2, const float* __restrict__ bm2,
    const float* __restrict__ Wma, const float* __restrict__ bma,
    const unsigned short* __restrict__ We1, const float* __restrict__ be1,
    const unsigned short* __restrict__ We2, const float* __restrict__ be2,
    float* __restrict__ m_aggr, float* __restrict__ e_out) {
  __shared__ unsigned short s_m[8][16 * HID];      // 32KB/block

  const int wave = threadIdx.x >> 5, lane = threadIdx.x & 31;
  const int r = lane & 15, hi = lane >> 4;
  const int tile = blockIdx.x * 8 + wave;          // EE/16 == 50000, exact grid

  const int erow = tile * 16 + r;
  const int src = ei[erow], dst = ei[EE + erow];
  const unsigned short* hsb = h_bf + (size_t)src * HD;
  const unsigned short* hdb = h_bf + (size_t)dst * HD;
  const float*          epf = e + (size_t)erow * ED;
  const unsigned short* gbb = g_bf + (size_t)batch[dst] * GD;

  unsigned short* sm = s_m[wave];

  // ---- layer 1: [16,384] @ [384,128], A+B double-buffered ---------------
  auto loadA1 = [&](int kt) -> Frag {
    if (kt < 4)  return afrag_bf(hsb, kt * 32 + hi * 8);
    if (kt < 8)  return afrag_bf(hdb, (kt - 4) * 32 + hi * 8);
    if (kt < 10) return afrag_f32(epf, (kt - 8) * 32 + hi * 8);
    return afrag_bf(gbb, (kt - 10) * 32 + hi * 8);
  };
  v8f acc[8];
#pragma unroll
  for (int i = 0; i < 8; i++) acc[i] = (v8f)0.f;
  Frag an = loadA1(0);
  Frag bn[8];
#pragma unroll
  for (int nt = 0; nt < 8; ++nt) bn[nt] = bfrag(Wm1, nt, lane);
#pragma unroll
  for (int kt = 0; kt < 12; ++kt) {
    Frag a = an;
    Frag bc[8];
#pragma unroll
    for (int nt = 0; nt < 8; ++nt) bc[nt] = bn[nt];
    if (kt < 11) {
      an = loadA1(kt + 1);
#pragma unroll
      for (int nt = 0; nt < 8; ++nt) bn[nt] = bfrag(Wm1, (kt + 1) * 8 + nt, lane);
    }
#pragma unroll
    for (int nt = 0; nt < 8; ++nt) acc[nt] = wmma_bf16(a, bc[nt], acc[nt]);
  }
#pragma unroll
  for (int nt = 0; nt < 8; ++nt) {
    const float bias = bm1[nt * 16 + r];
#pragma unroll
    for (int i = 0; i < 8; i++) {
      float v = fmaxf(acc[nt][i] + bias, 0.f);
      sm[(i + hi * 8) * HID + nt * 16 + r] = f2bf_bits(v);
    }
  }

  // ---- layer 2: [16,128] @ [128,128] -> m ------------------------------
#pragma unroll
  for (int i = 0; i < 8; i++) acc[i] = (v8f)0.f;
#pragma unroll
  for (int nt = 0; nt < 8; ++nt) bn[nt] = bfrag(Wm2, nt, lane);
#pragma unroll
  for (int kt = 0; kt < 4; ++kt) {
    Frag a = afrag_bf(sm + r * HID, kt * 32 + hi * 8);
    Frag bc[8];
#pragma unroll
    for (int nt = 0; nt < 8; ++nt) bc[nt] = bn[nt];
    if (kt < 3) {
#pragma unroll
      for (int nt = 0; nt < 8; ++nt) bn[nt] = bfrag(Wm2, (kt + 1) * 8 + nt, lane);
    }
#pragma unroll
    for (int nt = 0; nt < 8; ++nt) acc[nt] = wmma_bf16(a, bc[nt], acc[nt]);
  }
#pragma unroll
  for (int nt = 0; nt < 8; ++nt) {
    const float bias = bm2[nt * 16 + r];
#pragma unroll
    for (int i = 0; i < 8; i++) {
      float v = fmaxf(acc[nt][i] + bias, 0.f);
      acc[nt][i] = v;                                  // keep m in registers
      sm[(i + hi * 8) * HID + nt * 16 + r] = f2bf_bits(v);
    }
  }

  // ---- attention: sigmoid(m @ Wma + bma); broadcast via shuffle ---------
  float part = 0.f;
  for (int c = hi * 64; c < hi * 64 + 64; ++c)
    part += bf2f(sm[r * HID + c]) * Wma[c];
  float tot = part + __shfl_xor(part, 16, 32);
  float attv = 1.f / (1.f + __expf(-(tot + bma[0])));   // lane l holds att(row l&15)

  // ---- m_aggr[dst] += att * m  (f32 atomics) ---------------------------
#pragma unroll
  for (int i = 0; i < 8; i++) {
    const int row = i + hi * 8;
    const int d2 = ei[EE + tile * 16 + row];
    const float a2 = __shfl(attv, row, 32);
    float* dp = m_aggr + (size_t)d2 * HID;
#pragma unroll
    for (int nt = 0; nt < 8; ++nt)
      atomicAdd(dp + nt * 16 + r, a2 * acc[nt][i]);
  }

  // ---- e_update: relu(m@We1+b)@We2+b ; e_out = relu(e + e_update) -------
  v8f ae[4];
#pragma unroll
  for (int i = 0; i < 4; i++) ae[i] = (v8f)0.f;
  Frag en[4];
#pragma unroll
  for (int nt = 0; nt < 4; ++nt) en[nt] = bfrag(We1, nt, lane);
#pragma unroll
  for (int kt = 0; kt < 4; ++kt) {
    Frag a = afrag_bf(sm + r * HID, kt * 32 + hi * 8);
    Frag ec[4];
#pragma unroll
    for (int nt = 0; nt < 4; ++nt) ec[nt] = en[nt];
    if (kt < 3) {
#pragma unroll
      for (int nt = 0; nt < 4; ++nt) en[nt] = bfrag(We1, (kt + 1) * 4 + nt, lane);
    }
#pragma unroll
    for (int nt = 0; nt < 4; ++nt) ae[nt] = wmma_bf16(a, ec[nt], ae[nt]);
  }
  // t tile overwrites start of sm (m no longer needed)
#pragma unroll
  for (int nt = 0; nt < 4; ++nt) {
    const float bias = be1[nt * 16 + r];
#pragma unroll
    for (int i = 0; i < 8; i++)
      sm[(i + hi * 8) * ED + nt * 16 + r] = f2bf_bits(fmaxf(ae[nt][i] + bias, 0.f));
  }
#pragma unroll
  for (int i = 0; i < 4; i++) ae[i] = (v8f)0.f;
#pragma unroll
  for (int nt = 0; nt < 4; ++nt) en[nt] = bfrag(We2, nt, lane);
#pragma unroll
  for (int kt = 0; kt < 2; ++kt) {
    Frag a = afrag_bf(sm + r * ED, kt * 32 + hi * 8);
    Frag ec[4];
#pragma unroll
    for (int nt = 0; nt < 4; ++nt) ec[nt] = en[nt];
    if (kt < 1) {
#pragma unroll
      for (int nt = 0; nt < 4; ++nt) en[nt] = bfrag(We2, 4 + nt, lane);
    }
#pragma unroll
    for (int nt = 0; nt < 4; ++nt) ae[nt] = wmma_bf16(a, ec[nt], ae[nt]);
  }
#pragma unroll
  for (int nt = 0; nt < 4; ++nt) {
    const float bias = be2[nt * 16 + r];
#pragma unroll
    for (int i = 0; i < 8; i++) {
      const int row = i + hi * 8;
      const size_t off = (size_t)(tile * 16 + row) * ED + nt * 16 + r;
      e_out[off] = fmaxf(e[off] + ae[nt][i] + bias, 0.f);
    }
  }
}

// ---------------------------------------------------------------------------
// Node kernel: h_cat=[h|m_aggr|g[batch]] (K=320) -> MLP -> h_out=relu(h+upd),
// h_att = h_out@Wha+bha (linear), h_aggr[batch] += h_att*h_out
// ---------------------------------------------------------------------------
__global__ __launch_bounds__(256) void node_kernel(
    const float* __restrict__ h, const unsigned short* __restrict__ h_bf,
    const unsigned short* __restrict__ g_bf,
    const int* __restrict__ batch, const float* __restrict__ m_aggr,
    const unsigned short* __restrict__ Wh1, const float* __restrict__ bh1,
    const unsigned short* __restrict__ Wh2, const float* __restrict__ bh2,
    const float* __restrict__ Wha, const float* __restrict__ bha,
    float* __restrict__ h_aggr, float* __restrict__ h_out) {
  __shared__ unsigned short s_m[8][16 * HID];      // 32KB/block

  const int wave = threadIdx.x >> 5, lane = threadIdx.x & 31;
  const int r = lane & 15, hi = lane >> 4;
  const int tile = blockIdx.x * 8 + wave;
  if (tile >= NN / 16) return;                     // 3125 tiles

  const int node = tile * 16 + r;
  const unsigned short* hp = h_bf + (size_t)node * HD;
  const float*          mp = m_aggr + (size_t)node * HID;
  const unsigned short* gp = g_bf + (size_t)batch[node] * GD;
  unsigned short* sm = s_m[wave];

  auto loadA1 = [&](int kt) -> Frag {
    if (kt < 4) return afrag_bf(hp, kt * 32 + hi * 8);
    if (kt < 8) return afrag_f32(mp, (kt - 4) * 32 + hi * 8);
    return afrag_bf(gp, (kt - 8) * 32 + hi * 8);
  };
  v8f acc[8];
#pragma unroll
  for (int i = 0; i < 8; i++) acc[i] = (v8f)0.f;
  Frag an = loadA1(0);
  Frag bn[8];
#pragma unroll
  for (int nt = 0; nt < 8; ++nt) bn[nt] = bfrag(Wh1, nt, lane);
#pragma unroll
  for (int kt = 0; kt < 10; ++kt) {                // K = 320
    Frag a = an;
    Frag bc[8];
#pragma unroll
    for (int nt = 0; nt < 8; ++nt) bc[nt] = bn[nt];
    if (kt < 9) {
      an = loadA1(kt + 1);
#pragma unroll
      for (int nt = 0; nt < 8; ++nt) bn[nt] = bfrag(Wh1, (kt + 1) * 8 + nt, lane);
    }
#pragma unroll
    for (int nt = 0; nt < 8; ++nt) acc[nt] = wmma_bf16(a, bc[nt], acc[nt]);
  }
#pragma unroll
  for (int nt = 0; nt < 8; ++nt) {
    const float bias = bh1[nt * 16 + r];
#pragma unroll
    for (int i = 0; i < 8; i++)
      sm[(i + hi * 8) * HID + nt * 16 + r] = f2bf_bits(fmaxf(acc[nt][i] + bias, 0.f));
  }

#pragma unroll
  for (int i = 0; i < 8; i++) acc[i] = (v8f)0.f;
#pragma unroll
  for (int nt = 0; nt < 8; ++nt) bn[nt] = bfrag(Wh2, nt, lane);
#pragma unroll
  for (int kt = 0; kt < 4; ++kt) {
    Frag a = afrag_bf(sm + r * HID, kt * 32 + hi * 8);
    Frag bc[8];
#pragma unroll
    for (int nt = 0; nt < 8; ++nt) bc[nt] = bn[nt];
    if (kt < 3) {
#pragma unroll
      for (int nt = 0; nt < 8; ++nt) bn[nt] = bfrag(Wh2, (kt + 1) * 8 + nt, lane);
    }
#pragma unroll
    for (int nt = 0; nt < 8; ++nt) acc[nt] = wmma_bf16(a, bc[nt], acc[nt]);
  }
  // h_out = relu(h + upd + bh2); stash bf16 copy in LDS for attention dot
#pragma unroll
  for (int nt = 0; nt < 8; ++nt) {
    const float bias = bh2[nt * 16 + r];
#pragma unroll
    for (int i = 0; i < 8; i++) {
      const int row = i + hi * 8;
      const size_t off = (size_t)(tile * 16 + row) * HD + nt * 16 + r;
      float v = fmaxf(h[off] + acc[nt][i] + bias, 0.f);
      acc[nt][i] = v;
      h_out[off] = v;
      sm[row * HD + nt * 16 + r] = f2bf_bits(v);
    }
  }
  // h_att = h_out @ Wha + bha (linear)
  float part = 0.f;
  for (int c = hi * 64; c < hi * 64 + 64; ++c)
    part += bf2f(sm[r * HD + c]) * Wha[c];
  float attv = part + __shfl_xor(part, 16, 32) + bha[0];

#pragma unroll
  for (int i = 0; i < 8; i++) {
    const int row = i + hi * 8;
    const int b = batch[tile * 16 + row];
    const float a2 = __shfl(attv, row, 32);
    float* dp = h_aggr + (size_t)b * HD;
#pragma unroll
    for (int nt = 0; nt < 8; ++nt)
      atomicAdd(dp + nt * 16 + r, a2 * acc[nt][i]);
  }
}

// ---------------------------------------------------------------------------
// Graph kernel: g_cat=[g|h_aggr] (K=192) -> MLP -> g_out = relu(g + upd)
// ---------------------------------------------------------------------------
__global__ __launch_bounds__(256) void graph_kernel(
    const float* __restrict__ g, const unsigned short* __restrict__ g_bf,
    const float* __restrict__ h_aggr,
    const unsigned short* __restrict__ Wg1, const float* __restrict__ bg1,
    const unsigned short* __restrict__ Wg2, const float* __restrict__ bg2,
    float* __restrict__ g_out) {
  __shared__ unsigned short s_t[8][16 * GD];       // 16KB/block

  const int wave = threadIdx.x >> 5, lane = threadIdx.x & 31;
  const int r = lane & 15, hi = lane >> 4;
  const int tile = blockIdx.x * 8 + wave;
  if (tile >= GGG / 16) return;                    // 32 tiles

  const int gi = tile * 16 + r;
  const unsigned short* gp = g_bf + (size_t)gi * GD;
  const float*          ap = h_aggr + (size_t)gi * HD;
  unsigned short* st = s_t[wave];

  v8f acc[4];
#pragma unroll
  for (int i = 0; i < 4; i++) acc[i] = (v8f)0.f;
  Frag bn[4];
#pragma unroll
  for (int nt = 0; nt < 4; ++nt) bn[nt] = bfrag(Wg1, nt, lane);
#pragma unroll
  for (int kt = 0; kt < 6; ++kt) {                 // K = 192
    Frag a = (kt < 2) ? afrag_bf(gp, kt * 32 + hi * 8)
                      : afrag_f32(ap, (kt - 2) * 32 + hi * 8);
    Frag bc[4];
#pragma unroll
    for (int nt = 0; nt < 4; ++nt) bc[nt] = bn[nt];
    if (kt < 5) {
#pragma unroll
      for (int nt = 0; nt < 4; ++nt) bn[nt] = bfrag(Wg1, (kt + 1) * 4 + nt, lane);
    }
#pragma unroll
    for (int nt = 0; nt < 4; ++nt) acc[nt] = wmma_bf16(a, bc[nt], acc[nt]);
  }
#pragma unroll
  for (int nt = 0; nt < 4; ++nt) {
    const float bias = bg1[nt * 16 + r];
#pragma unroll
    for (int i = 0; i < 8; i++)
      st[(i + hi * 8) * GD + nt * 16 + r] = f2bf_bits(fmaxf(acc[nt][i] + bias, 0.f));
  }
#pragma unroll
  for (int i = 0; i < 4; i++) acc[i] = (v8f)0.f;
#pragma unroll
  for (int nt = 0; nt < 4; ++nt) bn[nt] = bfrag(Wg2, nt, lane);
#pragma unroll
  for (int kt = 0; kt < 2; ++kt) {
    Frag a = afrag_bf(st + r * GD, kt * 32 + hi * 8);
    Frag bc[4];
#pragma unroll
    for (int nt = 0; nt < 4; ++nt) bc[nt] = bn[nt];
    if (kt < 1) {
#pragma unroll
      for (int nt = 0; nt < 4; ++nt) bn[nt] = bfrag(Wg2, 4 + nt, lane);
    }
#pragma unroll
    for (int nt = 0; nt < 4; ++nt) acc[nt] = wmma_bf16(a, bc[nt], acc[nt]);
  }
#pragma unroll
  for (int nt = 0; nt < 4; ++nt) {
    const float bias = bg2[nt * 16 + r];
#pragma unroll
    for (int i = 0; i < 8; i++) {
      const int row = i + hi * 8;
      const size_t off = (size_t)(tile * 16 + row) * GD + nt * 16 + r;
      g_out[off] = fmaxf(g[off] + acc[nt][i] + bias, 0.f);
    }
  }
}

// ---------------------------------------------------------------------------
extern "C" void kernel_launch(void* const* d_in, const int* in_sizes, int n_in,
                              void* d_out, int out_size, void* d_ws, size_t ws_size,
                              hipStream_t stream) {
  const float* h     = (const float*)d_in[0];
  const int*   ei    = (const int*)d_in[1];
  const float* e     = (const float*)d_in[2];
  const float* g     = (const float*)d_in[3];
  const int*   batch = (const int*)d_in[4];
  const float* W_m1 = (const float*)d_in[5];  const float* b_m1 = (const float*)d_in[6];
  const float* W_m2 = (const float*)d_in[7];  const float* b_m2 = (const float*)d_in[8];
  const float* W_ma = (const float*)d_in[9];  const float* b_ma = (const float*)d_in[10];
  const float* W_e1 = (const float*)d_in[11]; const float* b_e1 = (const float*)d_in[12];
  const float* W_e2 = (const float*)d_in[13]; const float* b_e2 = (const float*)d_in[14];
  const float* W_h1 = (const float*)d_in[15]; const float* b_h1 = (const float*)d_in[16];
  const float* W_h2 = (const float*)d_in[17]; const float* b_h2 = (const float*)d_in[18];
  const float* W_ha = (const float*)d_in[19]; const float* b_ha = (const float*)d_in[20];
  const float* W_g1 = (const float*)d_in[21]; const float* b_g1 = (const float*)d_in[22];
  const float* W_g2 = (const float*)d_in[23]; const float* b_g2 = (const float*)d_in[24];

  float* out   = (float*)d_out;
  float* h_out = out;                               // [N, HD]
  float* e_out = out + (size_t)NN * HD;             // [E, ED]
  float* g_out = e_out + (size_t)EE * ED;           // [G, GD]

  // workspace layout
  float* m_aggr = (float*)d_ws;                     // [N, HID] f32
  float* h_aggr = m_aggr + (size_t)NN * HID;        // [G, HD]  f32
  unsigned short* h_bf = (unsigned short*)(h_aggr + (size_t)GGG * HD);  // [N, HD]
  unsigned short* g_bf = h_bf + (size_t)NN * HD;                        // [G, GD]
  unsigned short* pw   = g_bf + (size_t)GGG * GD;
  unsigned short* Wm1p = pw; pw += 384 * 128;
  unsigned short* Wm2p = pw; pw += 128 * 128;
  unsigned short* We1p = pw; pw += 128 * 64;
  unsigned short* We2p = pw; pw += 64 * 64;
  unsigned short* Wh1p = pw; pw += 320 * 128;
  unsigned short* Wh2p = pw; pw += 128 * 128;
  unsigned short* Wg1p = pw; pw += 192 * 64;
  unsigned short* Wg2p = pw; pw += 64 * 64;

  const long zn = (long)NN * HID + (long)GGG * HD;  // m_aggr + h_aggr contiguous
  zero_kernel<<<512, 256, 0, stream>>>(m_aggr, zn);
  cvt_bf16_kernel<<<512, 256, 0, stream>>>(h, h_bf, (long)NN * HD);
  cvt_bf16_kernel<<<32,  256, 0, stream>>>(g, g_bf, (long)GGG * GD);

  pack_kernel<<<(384 * 128 + 255) / 256, 256, 0, stream>>>(W_m1, Wm1p, 384, 128);
  pack_kernel<<<(128 * 128 + 255) / 256, 256, 0, stream>>>(W_m2, Wm2p, 128, 128);
  pack_kernel<<<(128 * 64  + 255) / 256, 256, 0, stream>>>(W_e1, We1p, 128, 64);
  pack_kernel<<<(64  * 64  + 255) / 256, 256, 0, stream>>>(W_e2, We2p, 64, 64);
  pack_kernel<<<(320 * 128 + 255) / 256, 256, 0, stream>>>(W_h1, Wh1p, 320, 128);
  pack_kernel<<<(128 * 128 + 255) / 256, 256, 0, stream>>>(W_h2, Wh2p, 128, 128);
  pack_kernel<<<(192 * 64  + 255) / 256, 256, 0, stream>>>(W_g1, Wg1p, 192, 64);
  pack_kernel<<<(64  * 64  + 255) / 256, 256, 0, stream>>>(W_g2, Wg2p, 64, 64);

  edge_kernel<<<EE / 16 / 8, 256, 0, stream>>>(h_bf, ei, e, g_bf, batch,
      Wm1p, b_m1, Wm2p, b_m2, W_ma, b_ma, We1p, b_e1, We2p, b_e2,
      m_aggr, e_out);

  node_kernel<<<(NN / 16 + 7) / 8, 256, 0, stream>>>(h, h_bf, g_bf, batch, m_aggr,
      Wh1p, b_h1, Wh2p, b_h2, W_ha, b_ha, h_aggr, h_out);

  graph_kernel<<<(GGG / 16) / 8, 256, 0, stream>>>(g, g_bf, h_aggr,
      Wg1p, b_g1, Wg2p, b_g2, g_out);
}
